// SelfAttention_88880053224070
// MI455X (gfx1250) — compile-verified, matrix operand order checked by reading, not run on hardware
//
#include <hip/hip_runtime.h>
#include <hip/hip_bf16.h>

// Problem constants (reference: B=4, S=2048, D=1024, H=16, HD=64)
#define Bb 4
#define Ss 2048
#define Dd 1024
#define Hh 16
#define HDd 64

typedef __attribute__((ext_vector_type(16))) __bf16 v16bf;
typedef __attribute__((ext_vector_type(8)))  __bf16 bf16x8;
typedef __attribute__((ext_vector_type(4)))  __bf16 bf16x4;
typedef __attribute__((ext_vector_type(8)))  float  v8f;
typedef __attribute__((ext_vector_type(4)))  unsigned u32x4;
typedef __attribute__((ext_vector_type(8)))  unsigned u32x8;

// Build a 16x32 bf16 WMMA fragment (A or B) from two contiguous 8-element
// (16-byte) segments, matching the CDNA5 16-bit matrix lane layout:
//   elems 0..7  <-> K = half*8 + i
//   elems 8..15 <-> K = 16 + half*8 + i
__device__ __forceinline__ v16bf load_frag16(const __bf16* p0, const __bf16* p1) {
    v16bf r;
    ((bf16x8*)&r)[0] = *(const bf16x8*)p0;
    ((bf16x8*)&r)[1] = *(const bf16x8*)p1;
    return r;
}

// ---------------------------------------------------------------------------
// Tensor Data Mover: async-load a 2D bf16 tile [tile_d1 x tile_d0] from a
// row-major tensor (row stride = stride0 elements) into LDS at lds_addr,
// adding pad_amount DWORDs of LDS padding after every pad_interval DWORDs.
// D# bit layout per CDNA5 ISA ch.8 (group0 128b, group1 256b, 2D => groups
// 2/3 omitted).  Issued wave-uniform; tracked with TENSORcnt.
// ---------------------------------------------------------------------------
__device__ __forceinline__ void tdm_load_2d_bf16(
    unsigned lds_addr, const __bf16* gptr,
    unsigned tensor_d0, unsigned tensor_d1, unsigned stride0,
    unsigned tile_d0, unsigned tile_d1,
    unsigned pad_interval_log, unsigned pad_amount_m1) {
    unsigned long long ga = (unsigned long long)(uintptr_t)gptr;
    u32x4 g0;
    g0[0] = 1u;                                   // count=1 (valid user D#)
    g0[1] = lds_addr;                             // LDS byte address
    g0[2] = (unsigned)ga;                         // global_addr[31:0]
    g0[3] = (unsigned)(ga >> 32) | 0x80000000u;   // addr[56:32] | type=2<<30
    u32x8 g1;
    g1[0] = (1u << 16)                            // data_size = 2 bytes
          | (1u << 20)                            // pad_enable
          | (pad_interval_log << 22)              // 2^(v+1) DWORDs between pads
          | (pad_amount_m1 << 25);                // (v+1) DWORDs of pad
    g1[1] = (tensor_d0 << 16);                    // [47:32]=atomic addr(0); dim0 lo
    g1[2] = (tensor_d0 >> 16) | (tensor_d1 << 16);
    g1[3] = (tensor_d1 >> 16) | (tile_d0 << 16);  // [127:112] tile_dim0
    g1[4] = tile_d1;                              // tile_dim1; tile_dim2=0
    g1[5] = stride0;                              // tensor_dim0_stride[31:0]
    g1[6] = 0;                                    // stride0 hi / stride1 lo
    g1[7] = 0;
    asm volatile("tensor_load_to_lds %0, %1" :: "s"(g0), "s"(g1) : "memory");
}

// ---------------------------------------------------------------------------
// GEMM: C[M,N] = A[M,K] * B[K,N].  A is bf16 row-major (staged into LDS by
// the Tensor Data Mover, padded rows); B is fp32, converted to bf16 and
// transposed into LDS by the threads.  fp32 WMMA accumulation.
// Block: 256 threads = 8 waves (4x2), tile 128x128, BK=32.
// ---------------------------------------------------------------------------
template <bool OUT_BF16>
__global__ __launch_bounds__(256)
void gemm_bf16_tn(const __bf16* __restrict__ A, const float* __restrict__ Bm,
                  void* __restrict__ Cout, int M, int N, int K) {
    constexpr int BM = 128, BN = 128, BK = 32;
    constexpr int LDP = BK + 8;                 // 40 elems = 80B rows (16B-aligned)
    __shared__ __align__(16) __bf16 As[BM][LDP];
    __shared__ __align__(16) __bf16 BsT[BN][LDP];   // B transposed: [n][k]

    const int bm = blockIdx.y * BM;
    const int bn = blockIdx.x * BN;
    const int tid  = threadIdx.x;
    const int lane = tid & 31;
    const int wave = tid >> 5;
    const int wm = wave >> 1;        // 0..3
    const int wn = wave & 1;         // 0..1
    const int lrow = lane & 15;
    const int half = lane >> 4;
    const unsigned ldsA = (unsigned)(uintptr_t)(void*)&As[0][0];

    v8f acc[2][4] = {};

    for (int k0 = 0; k0 < K; k0 += BK) {
        // ---- A tile via TDM: 128x32 bf16, +4 DWORD pad per 16-DWORD row ----
        if (wave == 0) {
            tdm_load_2d_bf16(ldsA, A + (size_t)bm * K + k0,
                             (unsigned)K, (unsigned)M, (unsigned)K,
                             BK, BM, /*2^(3+1)=16 DW*/ 3u, /*4 DW*/ 3u);
        }
        // ---- stage B tile (32x128 f32 -> bf16, transposed into LDS) ----
#pragma unroll
        for (int i = 0; i < 4; ++i) {
            int idx = tid + i * 256;
            int r  = idx >> 5;                   // k row 0..31
            int c4 = idx & 31;                   // 0..31 float4 along N
            float4 v = ((const float4*)(Bm + (size_t)(k0 + r) * N + bn))[c4];
            int c = c4 * 4;
            BsT[c + 0][r] = (__bf16)v.x; BsT[c + 1][r] = (__bf16)v.y;
            BsT[c + 2][r] = (__bf16)v.z; BsT[c + 3][r] = (__bf16)v.w;
        }
        if (k0 + BK < K)   // prefetch next B tile -> global_prefetch_b8
            __builtin_prefetch(Bm + (size_t)(k0 + BK + (tid >> 3)) * N + bn, 0, 1);
        if (wave == 0)
            __builtin_amdgcn_s_wait_tensorcnt(0);   // TDM done before barrier
        __syncthreads();

        v16bf af[2], bf[4];
#pragma unroll
        for (int mi = 0; mi < 2; ++mi) {
            const __bf16* rp = &As[wm * 32 + mi * 16 + lrow][0];
            af[mi] = load_frag16(rp + half * 8, rp + 16 + half * 8);
        }
#pragma unroll
        for (int ni = 0; ni < 4; ++ni) {
            const __bf16* cp = &BsT[wn * 64 + ni * 16 + lrow][0];
            bf[ni] = load_frag16(cp + half * 8, cp + 16 + half * 8);
        }
#pragma unroll
        for (int mi = 0; mi < 2; ++mi)
#pragma unroll
            for (int ni = 0; ni < 4; ++ni)
                acc[mi][ni] = __builtin_amdgcn_wmma_f32_16x16x32_bf16(
                    false, af[mi], false, bf[ni], (short)0, acc[mi][ni], false, false);
        __syncthreads();
    }

    // ---- epilogue: C/D layout row = r + 8*half, col = lrow ----
#pragma unroll
    for (int mi = 0; mi < 2; ++mi)
#pragma unroll
        for (int ni = 0; ni < 4; ++ni)
#pragma unroll
            for (int r = 0; r < 8; ++r) {
                size_t row = (size_t)(bm + wm * 32 + mi * 16 + r + 8 * half);
                size_t col = (size_t)(bn + wn * 64 + ni * 16 + lrow);
                float v = acc[mi][ni][r];
                if (OUT_BF16)
                    ((__bf16*)Cout)[row * N + col] = (__bf16)v;
                else
                    ((float*)Cout)[row * N + col] = v;
            }
}

// Elementwise fp32 -> bf16 (4 per thread, 8B stores)
__global__ __launch_bounds__(256)
void f32_to_bf16(const float* __restrict__ in, __bf16* __restrict__ out) {
    int t = blockIdx.x * 256 + threadIdx.x;
    float4 v = ((const float4*)in)[t];
    bf16x4 o;
    o[0] = (__bf16)v.x; o[1] = (__bf16)v.y; o[2] = (__bf16)v.z; o[3] = (__bf16)v.w;
    ((bf16x4*)out)[t] = o;
}

// ---------------------------------------------------------------------------
// RoPE + head split for Q and K: qkv bf16 [B*S, 3D] -> Q,K bf16 [B][H][S][64]
// ---------------------------------------------------------------------------
__global__ __launch_bounds__(256)
void rope_qk(const __bf16* __restrict__ qkv, const float* __restrict__ cosT,
             const float* __restrict__ sinT, __bf16* __restrict__ Q,
             __bf16* __restrict__ Ko) {
    int tid = blockIdx.x * 256 + threadIdx.x;    // (b, s, h, i)
    int i = tid & 31;
    int h = (tid >> 5) & (Hh - 1);
    int s = (tid >> 9) & (Ss - 1);
    int b = tid >> 20;
    const __bf16* row = qkv + (size_t)(b * Ss + s) * (3 * Dd);
    float c  = cosT[s * 32 + i];
    float sn = sinT[s * 32 + i];
    size_t ob = ((size_t)(b * Hh + h) * Ss + s) * HDd;

    float x1 = (float)row[h * HDd + 2 * i];
    float x2 = (float)row[h * HDd + 2 * i + 1];
    Q[ob + i]      = (__bf16)(x1 * c - x2 * sn);
    Q[ob + 32 + i] = (__bf16)(x1 * sn + x2 * c);

    x1 = (float)row[Dd + h * HDd + 2 * i];
    x2 = (float)row[Dd + h * HDd + 2 * i + 1];
    Ko[ob + i]      = (__bf16)(x1 * c - x2 * sn);
    Ko[ob + 32 + i] = (__bf16)(x1 * sn + x2 * c);
}

// V split + transpose: qkv bf16 -> Vt[b][h][hd][s] so the P*V WMMA B-fragment
// (elements along the key dim) becomes two contiguous 16-byte global loads.
__global__ __launch_bounds__(256)
void v_transpose(const __bf16* __restrict__ qkv, __bf16* __restrict__ Vt) {
    int tid = blockIdx.x * 256 + threadIdx.x;    // (b, h, d, s), s fastest
    int s = tid & (Ss - 1);
    int d = (tid >> 11) & (HDd - 1);
    int h = (tid >> 17) & (Hh - 1);
    int b = tid >> 21;
    Vt[tid] = qkv[(size_t)(b * Ss + s) * (3 * Dd) + 2 * Dd + h * HDd + d];
}

// ---------------------------------------------------------------------------
// Flash attention, one wave per (b, h, 16-row q tile).  Key loop split into
// full chunks (no causal compares) and diagonal chunks (masked).  Output is
// written as bf16 so the final projection GEMM can TDM-load it directly.
// ---------------------------------------------------------------------------
__global__ __launch_bounds__(32)
void attn_fwd(const __bf16* __restrict__ Q, const __bf16* __restrict__ Kh,
              const __bf16* __restrict__ Vt, __bf16* __restrict__ Out) {
    __shared__ __align__(16) __bf16 Plds[16][40];

    const int lane = threadIdx.x;
    const int lrow = lane & 15;
    const int half = lane >> 4;
    const int q0 = blockIdx.x * 16;
    const int h  = blockIdx.y;
    const int b  = blockIdx.z;

    const __bf16* Qb = Q  + (size_t)(b * Hh + h) * Ss * HDd;
    const __bf16* Kb = Kh + (size_t)(b * Hh + h) * Ss * HDd;
    const __bf16* Vb = Vt + (size_t)(b * Hh + h) * HDd * Ss;

    v16bf aq[2];
#pragma unroll
    for (int c = 0; c < 2; ++c) {
        const __bf16* rp = Qb + (size_t)(q0 + lrow) * HDd + 32 * c;
        aq[c] = load_frag16(rp + half * 8, rp + 16 + half * 8);
    }

    v8f o[4] = {};
    float m[8], l[8];
#pragma unroll
    for (int r = 0; r < 8; ++r) { m[r] = -1e30f; l[r] = 0.f; }

    auto process = [&](int kb, bool masked) {
        v8f s0 = {}, s1 = {};
#pragma unroll
        for (int c = 0; c < 2; ++c) {
            const __bf16* k0p = Kb + (size_t)(kb + lrow) * HDd + 32 * c;
            const __bf16* k1p = Kb + (size_t)(kb + 16 + lrow) * HDd + 32 * c;
            v16bf bk0 = load_frag16(k0p + half * 8, k0p + 16 + half * 8);
            v16bf bk1 = load_frag16(k1p + half * 8, k1p + 16 + half * 8);
            s0 = __builtin_amdgcn_wmma_f32_16x16x32_bf16(false, aq[c], false, bk0,
                                                         (short)0, s0, false, false);
            s1 = __builtin_amdgcn_wmma_f32_16x16x32_bf16(false, aq[c], false, bk1,
                                                         (short)0, s1, false, false);
        }
        float p0[8], p1[8];
#pragma unroll
        for (int r = 0; r < 8; ++r) {
            float v0 = s0[r] * 0.125f;                 // 1/sqrt(64)
            float v1 = s1[r] * 0.125f;
            if (masked) {                              // compile-time folded
                int qg = q0 + r + 8 * half;
                if (kb + lrow > qg)      v0 = -1e30f;
                if (kb + 16 + lrow > qg) v1 = -1e30f;
            }
            float mx = fmaxf(v0, v1);
#pragma unroll
            for (int d2 = 1; d2 < 16; d2 <<= 1)
                mx = fmaxf(mx, __shfl_xor(mx, d2, 32));
            float mn = fmaxf(m[r], mx);
            float alpha = __expf(m[r] - mn);
            p0[r] = __expf(v0 - mn);
            p1[r] = __expf(v1 - mn);
            float rs = p0[r] + p1[r];
#pragma unroll
            for (int d2 = 1; d2 < 16; d2 <<= 1)
                rs += __shfl_xor(rs, d2, 32);
            l[r] = l[r] * alpha + rs;
            m[r] = mn;
#pragma unroll
            for (int g = 0; g < 4; ++g) o[g][r] *= alpha;
        }
        __syncthreads();                               // single-wave: ~S_NOP
#pragma unroll
        for (int r = 0; r < 8; ++r) {
            Plds[r + 8 * half][lrow]      = (__bf16)p0[r];
            Plds[r + 8 * half][16 + lrow] = (__bf16)p1[r];
        }
        __syncthreads();
        const __bf16* pr = &Plds[lrow][0];
        v16bf ap = load_frag16(pr + half * 8, pr + 16 + half * 8);
#pragma unroll
        for (int g = 0; g < 4; ++g) {
            const __bf16* vp = Vb + (size_t)(g * 16 + lrow) * Ss + kb;
            v16bf bv = load_frag16(vp + half * 8, vp + 16 + half * 8);
            o[g] = __builtin_amdgcn_wmma_f32_16x16x32_bf16(false, ap, false, bv,
                                                           (short)0, o[g], false, false);
        }
    };

    const int nchunk = (q0 + 15) / 32 + 1;
    const int jfull  = (q0 + 1) / 32;        // chunks with kb+31 <= q0: unmasked
    for (int j = 0; j < jfull; ++j)  process(32 * j, false);
    for (int j = jfull; j < nchunk; ++j) process(32 * j, true);

    float inv[8];
#pragma unroll
    for (int r = 0; r < 8; ++r) inv[r] = 1.0f / l[r];
#pragma unroll
    for (int g = 0; g < 4; ++g)
#pragma unroll
        for (int r = 0; r < 8; ++r) {
            size_t row = (size_t)b * Ss + q0 + r + 8 * half;
            Out[row * Dd + h * HDd + g * 16 + lrow] = (__bf16)(o[g][r] * inv[r]);
        }
}

// ---------------------------------------------------------------------------
extern "C" void kernel_launch(void* const* d_in, const int* in_sizes, int n_in,
                              void* d_out, int out_size, void* d_ws, size_t ws_size,
                              hipStream_t stream) {
    (void)in_sizes; (void)n_in; (void)out_size; (void)ws_size;
    const float* x    = (const float*)d_in[0];
    const float* cosT = (const float*)d_in[1];
    const float* sinT = (const float*)d_in[2];
    // d_in[3] = mask: unused, causality is computed analytically
    const float* Wqkv = (const float*)d_in[4];
    const float* Wout = (const float*)d_in[5];

    // workspace layout (128 MB total)
    char* ws = (char*)d_ws;
    __bf16* qkv = (__bf16*)(ws);                           // B*S*3D bf16 = 48 MB
    __bf16* Qh  = (__bf16*)(ws + (size_t)50331648);        // 16 MB
    __bf16* Kh  = (__bf16*)(ws + (size_t)67108864);        // 16 MB
    __bf16* Vt  = (__bf16*)(ws + (size_t)83886080);        // 16 MB
    __bf16* aob = (__bf16*)(ws + (size_t)100663296);       // attn out bf16 = 16 MB
    __bf16* xb  = (__bf16*)(ws + (size_t)117440512);       // x bf16 = 16 MB
    float*  out = (float*)d_out;

    const int M = Bb * Ss;   // 8192

    // 0) x -> bf16 (TDM source for GEMM 1)
    f32_to_bf16<<<(Bb * Ss * Dd) / 1024, 256, 0, stream>>>(x, xb);
    // 1) qkv = x @ Wqkv  (bf16 out)
    gemm_bf16_tn<true><<<dim3((3 * Dd) / 128, M / 128), 256, 0, stream>>>(
        xb, Wqkv, (void*)qkv, M, 3 * Dd, Dd);
    // 2) RoPE + head split for q,k
    rope_qk<<<(Bb * Ss * Hh * 32) / 256, 256, 0, stream>>>(qkv, cosT, sinT, Qh, Kh);
    // 3) v head split + transpose
    v_transpose<<<(Bb * Ss * Dd) / 256, 256, 0, stream>>>(qkv, Vt);
    // 4) causal flash attention (bf16 out)
    attn_fwd<<<dim3(Ss / 16, Hh, Bb), 32, 0, stream>>>(Qh, Kh, Vt, aob);
    // 5) out = aob @ Wout  (fp32 out)
    gemm_bf16_tn<false><<<dim3(Dd / 128, M / 128), 256, 0, stream>>>(
        aob, Wout, (void*)out, M, Dd, Dd);
}
